// SpraseLayerWithConnection_86509231276657
// MI455X (gfx1250) — matrix operand order repaired and checked by reading, not compiled, not run on hardware
//
#include <hip/hip_runtime.h>
#include <hip/hip_bf16.h>

typedef __attribute__((ext_vector_type(16))) _Float16 v16h;
typedef __attribute__((ext_vector_type(8)))  float    v8f;

#define BATCH 128
#define N_IN  16384
#define N_OUT 8192
#define ECHUNK 256   // edges staged per barrier round (8 WMMA chunks of K=32)

// ---------------------------------------------------------------------------
// Kernel 1: transpose + f32->f16 convert:  xt[f][b] = (f16) x[b][f]
// x is [128, 16384] row-major; xt is [16384, 128] so each feature's batch
// column becomes a contiguous 256B row (coalesced WMMA A-matrix gathers).
// ---------------------------------------------------------------------------
__global__ __launch_bounds__(256) void transpose_to_f16(
    const float* __restrict__ x, _Float16* __restrict__ xt) {
  __shared__ float tile[BATCH][33];  // [b][f_local], padded vs bank conflicts
  const int f0  = blockIdx.x * 32;
  const int tid = threadIdx.x;
  const int fi  = tid & 31;   // 0..31
  const int b0  = tid >> 5;   // 0..7
  #pragma unroll
  for (int bb = 0; bb < BATCH; bb += 8) {
    tile[bb + b0][fi] = x[(bb + b0) * N_IN + (f0 + fi)];  // 128B coalesced
  }
  __syncthreads();
  const int f_local = tid >> 3;        // 0..31
  const int c0      = (tid & 7) * 16;  // 0,16,...,112
  #pragma unroll
  for (int k = 0; k < 16; ++k) {
    xt[(f0 + f_local) * BATCH + c0 + k] = (_Float16)tile[c0 + k][f_local];
  }
}

// ---------------------------------------------------------------------------
// Kernel 2: segmented gather-GEMM via v_wmma_f32_16x16x32_f16.
// Block: 16 consecutive outputs (N), 8 waves; wave w owns batch rows 16w..16w+15 (M).
// edge_dst is sorted -> this block's edges are one contiguous slice (binary
// search). Stage 256 edges per barrier round; 8 K=32 WMMA chunks per round.
// A[m,k] = xt[src_k][m];  B[k,n] = (dst_k == n) ? w_k : 0.
// ---------------------------------------------------------------------------
__global__ __launch_bounds__(256) void sparse_layer_wmma(
    const _Float16* __restrict__ xt,
    const float*    __restrict__ w,
    const float*    __restrict__ bias,
    const int*      __restrict__ esrc,
    const int*      __restrict__ edst,
    float*          __restrict__ y,
    int n_edges) {
  __shared__ int      s_src[ECHUNK];
  __shared__ unsigned s_dw[ECHUNK];      // (dst_local << 16) | f16bits(w)
  __shared__ int      s_bounds[2];
  __shared__ float    s_out[8][16][17];  // [wave][m][n] staging, padded

  const int n_base = blockIdx.x * 16;
  const int tid    = threadIdx.x;
  const int wv     = tid >> 5;   // wave id 0..7 -> batch tile
  const int lane   = tid & 31;   // wave32 lane

  // Segment bounds: lower_bound(edge_dst, n_base) / lower_bound(edge_dst, n_base+16)
  if (tid < 2) {
    const int target = n_base + (tid << 4);
    int lo = 0, hi = n_edges;
    while (lo < hi) {
      const int mid = (lo + hi) >> 1;
      if (edst[mid] < target) lo = mid + 1; else hi = mid;
    }
    s_bounds[tid] = lo;
  }
  __syncthreads();
  const int seg_lo = s_bounds[0];
  const int seg_hi = s_bounds[1];

  // Per-lane constants for the documented wave32 16-bit WMMA layouts.
  const int mrow    = (wv << 4) + (lane & 15);   // batch row of this lane's A elems
  const int kofsA   = (lane < 16) ? 0 : 8;       // A: K(i) = (i<8 ? i : i+8) + kofsA
  const int n_local = lane & 15;                 // B/C: N = lane % 16
  const int kofsB   = (lane < 16) ? 0 : 8;       // B: K(i) = (i<8 ? i : i+8) + kofsB

  v8f acc = {};  // 16x16 f32 accumulator tile

  for (int e0 = seg_lo; e0 < seg_hi; e0 += ECHUNK) {
    __syncthreads();  // all waves done reading previous round's LDS
    {
      const int e = e0 + tid;
      int      src = 0;
      unsigned dw  = 0x7FFFu << 16;  // dst=0x7FFF never matches n_local -> B=0
      if (e < seg_hi) {
        src = esrc[e];
        const short wb = __builtin_bit_cast(short, (_Float16)w[e]);
        dw = (((unsigned)(edst[e] - n_base)) << 16) | (unsigned short)wb;
      }
      s_src[tid] = src;
      s_dw[tid]  = dw;
      // Prefetch next round's edge lists while this round computes.
      const int ep = e + ECHUNK;
      if (ep < seg_hi) {
        __builtin_prefetch(&esrc[ep], 0, 1);
        __builtin_prefetch(&edst[ep], 0, 1);
        __builtin_prefetch(&w[ep],    0, 1);
      }
    }
    __syncthreads();

    const int nloc = min(ECHUNK, seg_hi - e0);  // block-uniform
    for (int c = 0; c < nloc; c += 32) {
      // A-matrix (16x32 f16): element i -> VGPR i/2, half i%2;
      // K(i,lane) = (i<8 ? i : i+8) + kofsA. 32B-coalesced reads of xt rows.
      v16h A, Bm;
      #pragma unroll
      for (int i = 0; i < 16; ++i) {
        const int kA = c + ((i < 8) ? i : i + 8) + kofsA;
        A[i] = xt[s_src[kA] * BATCH + mrow];
      }
      // B-matrix (32x16 f16): lane holds column n_local; element i -> K = (i<8 ? i : i+8) + kofsB.
      #pragma unroll
      for (int i = 0; i < 16; ++i) {
        const unsigned dw = s_dw[c + ((i < 8) ? i : i + 8) + kofsB];
        const short bits = ((int)(dw >> 16) == n_local) ? (short)(dw & 0xFFFFu)
                                                        : (short)0;
        Bm[i] = __builtin_bit_cast(_Float16, bits);
      }
      acc = __builtin_amdgcn_wmma_f32_16x16x32_f16(
          /*neg_a=*/false, A, /*neg_b=*/false, Bm,
          /*c_mod=*/(short)0, acc, /*reuse_a=*/false, /*reuse_b=*/false);
    }
  }

  // Epilogue: C layout -> lane,VGPR r holds (M = r + (lane<16?0:8), N = lane%16)
  const float bval = bias[n_base + n_local];
  const int   mhi  = (lane < 16) ? 0 : 8;
  #pragma unroll
  for (int r = 0; r < 8; ++r) {
    float v = acc[r] + bval;
    s_out[wv][r + mhi][n_local] = (v > 0.f) ? v : 0.f;
  }
  __syncthreads();

  // Coalesced store: 128 rows x 16 floats; 2 threads per row, 32B each.
  const int row = tid >> 1;       // batch index 0..127
  const int c0  = (tid & 1) * 8;  // 0 or 8
  float4* dst = (float4*)&y[row * N_OUT + n_base + c0];
  const float* src = &s_out[row >> 4][row & 15][c0];
  dst[0] = make_float4(src[0], src[1], src[2], src[3]);
  dst[1] = make_float4(src[4], src[5], src[6], src[7]);
}

// ---------------------------------------------------------------------------
extern "C" void kernel_launch(void* const* d_in, const int* in_sizes, int n_in,
                              void* d_out, int out_size, void* d_ws, size_t ws_size,
                              hipStream_t stream) {
  const float* x    = (const float*)d_in[0];
  const float* w    = (const float*)d_in[1];
  const float* bias = (const float*)d_in[2];
  const int*   esrc = (const int*)d_in[3];
  const int*   edst = (const int*)d_in[4];
  float*       y    = (float*)d_out;
  const int n_edges = in_sizes[3];

  _Float16* xt = (_Float16*)d_ws;  // 16384 * 128 * 2B = 4 MB scratch

  transpose_to_f16<<<N_IN / 32, 256, 0, stream>>>(x, xt);
  sparse_layer_wmma<<<N_OUT / 16, 256, 0, stream>>>(xt, w, bias, esrc, edst, y, n_edges);
}
